// Speller_65214783423127
// MI455X (gfx1250) — compile-verified, matrix operand order checked by reading, not run on hardware
//
#include <hip/hip_runtime.h>
#include <hip/hip_bf16.h>
#include <stdint.h>

// MI455X / gfx1250: wave32, WMMA 16x16x32 bf16 -> f32 accumulate.
typedef __attribute__((ext_vector_type(16))) __bf16 v16bf;
typedef __attribute__((ext_vector_type(8)))  __bf16 v8bf;
typedef __attribute__((ext_vector_type(8)))  float  v8f;

static constexpr int Sn = 512;   // source length
static constexpr int Bn = 512;   // batch
static constexpr int Tn = 160;   // decode steps
static constexpr int En = 256;   // embed / listener dim
static constexpr int Hn = 512;   // hidden
static constexpr int Vn = 34;    // vocab

static constexpr int LDT = 80;   // padded LDS row (elems): 160 B, multiple of 16 B

__device__ __forceinline__ float sigm(float x) { return 1.0f / (1.0f + __expf(-x)); }

// Build a v16bf WMMA fragment from two contiguous 8-bf16 runs in LDS.
// Per CDNA5 ISA bf16 layouts, lane (g = lane>>4) needs K = {koff+8g .. +7} and
// {koff+16+8g .. +7}: both are 16-byte ds_load_b128s.
__device__ __forceinline__ v16bf frag(const __bf16* rowp, int koff) {
  v8bf lo = *(const v8bf*)(rowp + koff);
  v8bf hi = *(const v8bf*)(rowp + koff + 16);
  return __builtin_shufflevector(lo, hi, 0, 1, 2, 3, 4, 5, 6, 7,
                                 8, 9, 10, 11, 12, 13, 14, 15);
}

// ---------------------------------------------------------------------------
// GEMM (TN):  C[M x N](f32) = A[M x K](bf16, row stride lda) * Wt[N x K](bf16)^T
// Block = 256 threads = 8 waves; block tile 32(M) x 128(N); K chunk = 64.
// Each wave owns a 16x32 slab: two f32 accumulators, A fragment reused.
// ---------------------------------------------------------------------------
__global__ __launch_bounds__(256) void k_gemm(const __bf16* __restrict__ A, int lda,
                                              const __bf16* __restrict__ Wt,
                                              float* __restrict__ C,
                                              int N, int K) {
  __shared__ __align__(16) __bf16 As[32][LDT];    // [m][k]
  __shared__ __align__(16) __bf16 Ws[128][LDT];   // [n][k] (already transposed)
  const int tid  = threadIdx.x;
  const int wave = tid >> 5, lane = tid & 31;
  const int mw = wave >> 2, nw = wave & 3;
  const int g = lane >> 4, li = lane & 15;
  const int row0 = blockIdx.x * 32, col0 = blockIdx.y * 128;
  v8f acc0 = {0.f, 0.f, 0.f, 0.f, 0.f, 0.f, 0.f, 0.f};
  v8f acc1 = {0.f, 0.f, 0.f, 0.f, 0.f, 0.f, 0.f, 0.f};

  for (int k0 = 0; k0 < K; k0 += 64) {
    // stage A: 32x64 bf16 = 256 octets -> one b128 per thread
    {
      int r = tid >> 3, c = (tid & 7) * 8;
      *(v8bf*)&As[r][c] = *(const v8bf*)&A[(size_t)(row0 + r) * lda + k0 + c];
    }
    // stage Wt: 128x64 bf16 = 1024 octets -> four b128 per thread
#pragma unroll
    for (int j = 0; j < 4; ++j) {
      int o = tid + 256 * j;
      int r = o >> 3, c = (o & 7) * 8;
      *(v8bf*)&Ws[r][c] = *(const v8bf*)&Wt[(size_t)(col0 + r) * K + k0 + c];
    }
    __syncthreads();

#pragma unroll
    for (int ks = 0; ks < 2; ++ks) {
      const int kb = 32 * ks + 8 * g;
      v16bf af = frag(&As[16 * mw + li][0], kb);
      v16bf b0 = frag(&Ws[32 * nw + li][0], kb);
      v16bf b1 = frag(&Ws[32 * nw + 16 + li][0], kb);
      acc0 = __builtin_amdgcn_wmma_f32_16x16x32_bf16(false, af, false, b0,
                                                     (short)0, acc0, false, false);
      acc1 = __builtin_amdgcn_wmma_f32_16x16x32_bf16(false, af, false, b1,
                                                     (short)0, acc1, false, false);
    }
    __syncthreads();
  }

  // C/D layout: lane holds col n = lane&15; VGPR r -> row m = r + 8*(lane>>4)
  const int n_ = lane & 15, half = lane >> 4;
#pragma unroll
  for (int r = 0; r < 8; ++r) {
    int m = r + 8 * half;
    float* crow = &C[(size_t)(row0 + 16 * mw + m) * N];
    crow[col0 + 32 * nw + n_]      = acc0[r];
    crow[col0 + 32 * nw + 16 + n_] = acc1[r];
  }
}

// ---------------------------------------------------------------------------
// One-time packing / conversion kernels (weights stored as [N][K] = torch-native)
// ---------------------------------------------------------------------------
// [Wih | Whh] (each 2048x512 row-major) -> Wt[n][k] bf16, N=2048, K=1024
__global__ void k_pack_lstm(const float* __restrict__ wih, const float* __restrict__ whh,
                            __bf16* __restrict__ dst) {
  for (int idx = blockIdx.x * blockDim.x + threadIdx.x; idx < 2048 * 1024;
       idx += gridDim.x * blockDim.x) {
    int n = idx >> 10, k = idx & 1023;
    float v = (k < Hn) ? wih[n * Hn + k] : whh[n * Hn + (k - Hn)];
    dst[idx] = (__bf16)v;
  }
}

// Wq (E x H) row-major is already [n][k]: direct convert
__global__ void k_pack_wq(const float* __restrict__ Wq, __bf16* __restrict__ dst) {
  for (int idx = blockIdx.x * blockDim.x + threadIdx.x; idx < En * Hn;
       idx += gridDim.x * blockDim.x) {
    dst[idx] = (__bf16)Wq[idx];
  }
}

// Wsc (V x 768) -> [n][k] bf16 with N padded 34 -> 128 (zeros)
__global__ void k_pack_wsc(const float* __restrict__ Wsc, __bf16* __restrict__ dst) {
  for (int idx = blockIdx.x * blockDim.x + threadIdx.x; idx < 128 * 768;
       idx += gridDim.x * blockDim.x) {
    int n = idx / 768, k = idx % 768;
    dst[idx] = (__bf16)((n < Vn) ? Wsc[n * 768 + k] : 0.0f);
  }
}

// key/value (S,B,E) f32 -> (B,S,E) bf16  (halves attention streaming bytes)
__global__ void k_cvt_kv(const float* __restrict__ src, __bf16* __restrict__ dst) {
  const int total = Sn * Bn * En;
  for (int idx = blockIdx.x * blockDim.x + threadIdx.x; idx < total;
       idx += gridDim.x * blockDim.x) {
    int s = idx >> 17;                 // B*E = 131072 = 2^17
    int rem = idx & 131071;
    int b = rem >> 8, e = rem & 255;
    dst[((size_t)b << 17) + ((size_t)s << 8) + e] = (__bf16)src[idx];
  }
}

// init c0,c1 (f32) and x1cat = [h0b | h1b] (bf16) from broadcast state vectors
__global__ void k_init(const float* __restrict__ s0, const float* __restrict__ s1,
                       const float* __restrict__ cs0, const float* __restrict__ cs1,
                       float* __restrict__ c0, float* __restrict__ c1,
                       __bf16* __restrict__ x1) {
  int idx = blockIdx.x * blockDim.x + threadIdx.x;   // exactly B*H threads
  int b = idx >> 9, j = idx & 511;
  c0[idx] = cs0[j];
  c1[idx] = cs1[j];
  x1[b * 1024 + j] = (__bf16)s0[j];
  x1[b * 1024 + 512 + j] = (__bf16)s1[j];
}

// ---------------------------------------------------------------------------
// Attention (batched matvec: memory-bound, wide global_load_b128 streams)
// ---------------------------------------------------------------------------
__global__ __launch_bounds__(256) void k_scores(const __bf16* __restrict__ kb,
                                                const float* __restrict__ q,
                                                const float* __restrict__ bq,
                                                float* __restrict__ e) {
  __shared__ float qs[En];
  const int b = blockIdx.x;
  const int s = blockIdx.y * 256 + threadIdx.x;
  qs[threadIdx.x] = q[b * En + threadIdx.x] + bq[threadIdx.x];
  __syncthreads();
  const v8bf* kr = (const v8bf*)(kb + ((size_t)b * Sn + s) * En);
  float acc = 0.f;
#pragma unroll 4
  for (int i = 0; i < En / 8; ++i) {
    v8bf kv = kr[i];
#pragma unroll
    for (int j = 0; j < 8; ++j) acc += (float)kv[j] * qs[i * 8 + j];
  }
  e[b * Sn + s] = acc;
}

// softmax over S, *mask, renormalize (eps=1e-12), then ctx = alpha^T V
__global__ __launch_bounds__(256) void k_softmax_ctx(const float* __restrict__ e,
                                                     const float* __restrict__ mask,
                                                     const __bf16* __restrict__ vb,
                                                     float* __restrict__ ctx,
                                                     __bf16* __restrict__ ctxb) {
  __shared__ float red[256];
  __shared__ float al[Sn];
  const int b = blockIdx.x, tid = threadIdx.x;
  float v0 = e[b * Sn + tid], v1 = e[b * Sn + 256 + tid];

  red[tid] = fmaxf(v0, v1);
  __syncthreads();
  for (int off = 128; off > 0; off >>= 1) {
    if (tid < off) red[tid] = fmaxf(red[tid], red[tid + off]);
    __syncthreads();
  }
  const float M = red[0];
  __syncthreads();

  float p0 = __expf(v0 - M), p1 = __expf(v1 - M);
  red[tid] = p0 + p1;
  __syncthreads();
  for (int off = 128; off > 0; off >>= 1) {
    if (tid < off) red[tid] += red[tid + off];
    __syncthreads();
  }
  const float psum = red[0];
  __syncthreads();

  float a0 = (p0 / psum) * mask[b * Sn + tid];
  float a1 = (p1 / psum) * mask[b * Sn + 256 + tid];
  red[tid] = a0 + a1;
  __syncthreads();
  for (int off = 128; off > 0; off >>= 1) {
    if (tid < off) red[tid] += red[tid + off];
    __syncthreads();
  }
  const float den = fmaxf(red[0], 1e-12f);
  al[tid] = a0 / den;
  al[tid + 256] = a1 / den;
  __syncthreads();

  // ctx[b, e]: one thread per e, coalesced V reads, alpha broadcast from LDS
  float acc = 0.f;
  const __bf16* vrow = vb + ((size_t)b * Sn) * En + tid;
  for (int s = 0; s < Sn; ++s) {
    if (s + 8 < Sn) __builtin_prefetch(vrow + (size_t)(s + 8) * En, 0, 0);
    acc += al[s] * (float)vrow[(size_t)s * En];
  }
  ctx[b * En + tid] = acc;
  ctxb[b * En + tid] = (__bf16)acc;
}

// ---------------------------------------------------------------------------
// Per-step glue kernels
// ---------------------------------------------------------------------------
// x0cat = [Emb[label_t], ctx, h0] as bf16  (B x 1024)
__global__ void k_build_x0(const int* __restrict__ labels, const float* __restrict__ Emb,
                           const float* __restrict__ ctx, const __bf16* __restrict__ x1,
                           __bf16* __restrict__ x0, int t) {
  for (int idx = blockIdx.x * blockDim.x + threadIdx.x; idx < Bn * 1024;
       idx += gridDim.x * blockDim.x) {
    int b = idx >> 10, j = idx & 1023;
    if (j < 256) {
      int lab = labels[b * Tn + t];
      x0[idx] = (__bf16)Emb[lab * En + j];
    } else if (j < 512) {
      x0[idx] = (__bf16)ctx[b * En + (j - 256)];
    } else {
      x0[idx] = x1[b * 1024 + (j - 512)];   // h0 (bf16)
    }
  }
}

// LSTM pointwise: gate order i,f,g,o ; c in-place ; writes h as bf16 into x1cat
__global__ void k_pointwise(const float* __restrict__ gates,
                            const float* __restrict__ bih, const float* __restrict__ bhh,
                            float* __restrict__ c, __bf16* __restrict__ hdst, int hoff) {
  int idx = blockIdx.x * blockDim.x + threadIdx.x;   // exactly B*H threads
  int b = idx >> 9, j = idx & 511;
  const float* g = gates + (size_t)b * 2048;
  float gi = g[j]        + bih[j]        + bhh[j];
  float gf = g[512 + j]  + bih[512 + j]  + bhh[512 + j];
  float gg = g[1024 + j] + bih[1024 + j] + bhh[1024 + j];
  float go = g[1536 + j] + bih[1536 + j] + bhh[1536 + j];
  float cn = sigm(gf) * c[idx] + sigm(gi) * tanhf(gg);
  float hn = sigm(go) * tanhf(cn);
  c[idx] = cn;
  hdst[b * 1024 + hoff + j] = (__bf16)hn;
}

// scat = [h1 (bf16 from x1cat tail), ctx (bf16)]  (B x 768)
__global__ void k_build_sc(const __bf16* __restrict__ x1, const __bf16* __restrict__ ctxb,
                           __bf16* __restrict__ scat) {
  for (int idx = blockIdx.x * blockDim.x + threadIdx.x; idx < Bn * 768;
       idx += gridDim.x * blockDim.x) {
    int b = idx / 768, j = idx % 768;
    scat[idx] = (j < 512) ? x1[b * 1024 + 512 + j] : ctxb[b * En + (j - 512)];
  }
}

// out[b, t, v] = otmp[b, v] + bsc[v]
__global__ void k_out_bias(const float* __restrict__ otmp, const float* __restrict__ bsc,
                           float* __restrict__ out, int t) {
  int idx = blockIdx.x * blockDim.x + threadIdx.x;   // exactly B*V threads
  int b = idx / Vn, v = idx % Vn;
  out[(size_t)b * Tn * Vn + (size_t)t * Vn + v] = otmp[b * 128 + v] + bsc[v];
}

// ---------------------------------------------------------------------------
extern "C" void kernel_launch(void* const* d_in, const int* in_sizes, int n_in,
                              void* d_out, int out_size, void* d_ws, size_t ws_size,
                              hipStream_t stream) {
  (void)in_sizes; (void)n_in; (void)out_size; (void)ws_size;
  const float* key    = (const float*)d_in[0];
  const float* value  = (const float*)d_in[1];
  const float* mask   = (const float*)d_in[2];
  const int*   labels = (const int*)d_in[3];
  const float* Wq     = (const float*)d_in[4];
  const float* bq     = (const float*)d_in[5];
  const float* Wsc    = (const float*)d_in[6];
  const float* bsc    = (const float*)d_in[7];
  const float* Emb    = (const float*)d_in[8];
  const float* Wih0   = (const float*)d_in[9];
  const float* Whh0   = (const float*)d_in[10];
  const float* bih0   = (const float*)d_in[11];
  const float* bhh0   = (const float*)d_in[12];
  const float* Wih1   = (const float*)d_in[13];
  const float* Whh1   = (const float*)d_in[14];
  const float* bih1   = (const float*)d_in[15];
  const float* bhh1   = (const float*)d_in[16];
  const float* s0     = (const float*)d_in[17];
  const float* s1     = (const float*)d_in[18];
  const float* cs0    = (const float*)d_in[19];
  const float* cs1    = (const float*)d_in[20];
  float* out = (float*)d_out;

  // workspace carving (bump allocator, 256B aligned)
  char* p = (char*)d_ws;
  auto alloc = [&](size_t n) -> char* {
    char* r = p; p += (n + 255) & ~(size_t)255; return r;
  };
  __bf16* w0   = (__bf16*)alloc(2ull * 2048 * 1024);  // [N=2048][K=1024]
  __bf16* w1   = (__bf16*)alloc(2ull * 2048 * 1024);
  __bf16* wq   = (__bf16*)alloc(2ull * En * Hn);      // [N=256][K=512]
  __bf16* wsc  = (__bf16*)alloc(2ull * 128 * 768);    // [N=128 pad][K=768]
  __bf16* kb   = (__bf16*)alloc(2ull * Bn * Sn * En);
  __bf16* vb   = (__bf16*)alloc(2ull * Bn * Sn * En);
  __bf16* x0   = (__bf16*)alloc(2ull * Bn * 1024);
  __bf16* x1   = (__bf16*)alloc(2ull * Bn * 1024);    // [h0b | h1b]
  __bf16* scat = (__bf16*)alloc(2ull * Bn * 768);
  __bf16* ctxb = (__bf16*)alloc(2ull * Bn * En);
  float* c0    = (float*)alloc(4ull * Bn * Hn);
  float* c1    = (float*)alloc(4ull * Bn * Hn);
  float* gates = (float*)alloc(4ull * Bn * 2048);
  float* q     = (float*)alloc(4ull * Bn * En);
  float* e     = (float*)alloc(4ull * Bn * Sn);
  float* ctx   = (float*)alloc(4ull * Bn * En);
  float* otmp  = (float*)alloc(4ull * Bn * 128);

  // one-time setup (deterministic, re-done every call)
  k_pack_lstm<<<dim3(2048), 256, 0, stream>>>(Wih0, Whh0, w0);
  k_pack_lstm<<<dim3(2048), 256, 0, stream>>>(Wih1, Whh1, w1);
  k_pack_wq  <<<dim3(512),  256, 0, stream>>>(Wq, wq);
  k_pack_wsc <<<dim3(384),  256, 0, stream>>>(Wsc, wsc);
  k_cvt_kv   <<<dim3(4096), 256, 0, stream>>>(key, kb);
  k_cvt_kv   <<<dim3(4096), 256, 0, stream>>>(value, vb);
  k_init     <<<dim3(1024), 256, 0, stream>>>(s0, s1, cs0, cs1, c0, c1, x1);

  auto attend = [&]() {
    // q = h1 @ WqT : M=512, N=256, K=512 (WMMA)
    k_gemm<<<dim3(16, 2), 256, 0, stream>>>(x1 + 512, 1024, wq, q, En, Hn);
    k_scores<<<dim3(Bn, 2), 256, 0, stream>>>(kb, q, bq, e);
    k_softmax_ctx<<<dim3(Bn), 256, 0, stream>>>(e, mask, vb, ctx, ctxb);
  };
  auto cells = [&](int t) {
    k_build_x0<<<dim3(2048), 256, 0, stream>>>(labels, Emb, ctx, x1, x0, t);
    // layer0 gates: M=512, N=2048, K=1024 (WMMA)
    k_gemm<<<dim3(16, 16), 256, 0, stream>>>(x0, 1024, w0, gates, 2048, 1024);
    k_pointwise<<<dim3(1024), 256, 0, stream>>>(gates, bih0, bhh0, c0, x1, 0);
    // layer1 gates: A = [h0_new | h1_old]
    k_gemm<<<dim3(16, 16), 256, 0, stream>>>(x1, 1024, w1, gates, 2048, 1024);
    k_pointwise<<<dim3(1024), 256, 0, stream>>>(gates, bih1, bhh1, c1, x1, 512);
  };
  auto score = [&](int t) {
    k_build_sc<<<dim3(1536), 256, 0, stream>>>(x1, ctxb, scat);
    // out: M=512, N=128(pad of 34), K=768 (WMMA)
    k_gemm<<<dim3(16, 1), 256, 0, stream>>>(scat, 768, wsc, otmp, 128, 768);
    k_out_bias<<<dim3(68), 256, 0, stream>>>(otmp, bsc, out, t);
  };

  // step 0: attend + cells, no output
  attend();
  cells(0);
  // steps 1..T-1: attend -> score (uses pre-update h1) -> cells
  for (int t = 1; t < Tn; ++t) {
    attend();
    score(t - 1);
    cells(t);
  }
  // final score
  attend();
  score(Tn - 1);
}
